// ProtoTree_18923625906464
// MI455X (gfx1250) — matrix operand order repaired and checked by reading, compiled, not run on hardware
//
#include <hip/hip_runtime.h>
#include <hip/hip_bf16.h>
#include <stdint.h>

// ---------------------------------------------------------------------------
// ProtoTree fused pipeline for MI455X (gfx1250, wave32, WMMA).
// Dominant cost: 100352 x 511 x 256 GEMM -> v_wmma_f32_16x16x32_f16.
// v3: 8-wave workgroups share the B panel through LDS, staged with
//     global_load_async_to_lds_b128 (ASYNCcnt) and double-buffered, so WMMA
//     issue is decoupled from global-load latency. Each wave owns 32 protos.
// ---------------------------------------------------------------------------

#define BATCH   128
#define DCH     256
#define WHPIX   784          // 28*28, = 49 * 16 exactly
#define NPROTO  511
#define PPAD    512
#define NLEAF   512
#define NCLS    200
#define DEPTH   9
#define NPTILES 49           // pixel tiles of 16
#define NCHUNK  2            // pixel-tile chunks (25 + 24)

#define LROW    264          // LDS panel row stride in halves (528 B) -> bank-conflict-free
#define PANELH  (16 * LROW)  // halves per panel buffer

typedef __attribute__((ext_vector_type(16))) _Float16 v16h;
typedef __attribute__((ext_vector_type(8)))  _Float16 v8h;
typedef __attribute__((ext_vector_type(8)))  float    v8f;

union AFrag { v16h v; v8h h[2]; };

// ---------------------------------------------------------------------------
// Kernel 0: prototypes fp32 (511x256) -> fp16 padded (512x256), plus p2[512].
// ---------------------------------------------------------------------------
__global__ void k_protos(const float* __restrict__ prot, _Float16* __restrict__ p16,
                         float* __restrict__ p2) {
    __shared__ float red[256];
    int p = blockIdx.x;
    int d = threadIdx.x;
    float v = (p < NPROTO) ? prot[p * DCH + d] : 0.0f;
    p16[p * DCH + d] = (_Float16)v;
    red[d] = v * v;
    __syncthreads();
    for (int s = 128; s > 0; s >>= 1) {
        if (d < s) red[d] += red[d + s];
        __syncthreads();
    }
    if (d == 0) p2[p] = red[0];
}

// ---------------------------------------------------------------------------
// Kernel 1: x2[b][pix] = sum_d xs[b][d][pix]^2  (coalesced across pix).
// ---------------------------------------------------------------------------
__global__ void k_x2(const float* __restrict__ xs, float* __restrict__ x2) {
    int pix = blockIdx.x * 256 + threadIdx.x;
    int b   = blockIdx.y;
    if (pix >= WHPIX) return;
    const float* row = xs + (size_t)b * DCH * WHPIX + pix;
    float acc = 0.0f;
    #pragma unroll 8
    for (int d = 0; d < DCH; ++d) {
        float v = row[(size_t)d * WHPIX];
        acc += v * v;
    }
    x2[b * WHPIX + pix] = acc;
}

// ---------------------------------------------------------------------------
// Kernel 2: transpose+convert xs (B,D,WH) f32 -> xsT (B,WH,D) f16 via LDS tile.
// ---------------------------------------------------------------------------
__global__ void k_transpose(const float* __restrict__ xs, _Float16* __restrict__ xsT) {
    __shared__ _Float16 tile[32][33];
    int tx = threadIdx.x, ty = threadIdx.y;
    int pix0 = blockIdx.x * 32;
    int d0   = blockIdx.y * 32;
    int b    = blockIdx.z;
    int pix  = pix0 + tx;
    #pragma unroll
    for (int i = 0; i < 4; ++i) {
        int d = d0 + ty + 8 * i;
        float v = (pix < WHPIX) ? xs[((size_t)b * DCH + d) * WHPIX + pix] : 0.0f;
        tile[ty + 8 * i][tx] = (_Float16)v;
    }
    __syncthreads();
    #pragma unroll
    for (int i = 0; i < 4; ++i) {
        int pw = pix0 + ty + 8 * i;
        if (pw < WHPIX)
            xsT[((size_t)b * WHPIX + pw) * DCH + d0 + tx] = tile[tx][ty + 8 * i];
    }
}

// ---------------------------------------------------------------------------
// Kernel 3 (main): fused distance GEMM + min over pixels.
// Block = 256 threads (8 waves). Wave w owns protos [m0 + 32w, m0 + 32w + 32).
// Grid = (2 proto-halves, 2 pixel-chunks, 128 b) -> 512 WGs = 4096 waves.
// B panel (16 pixels x 256 ch, f16) staged in LDS via async DMA, double-buffered.
//
// WMMA fragment layouts (CDNA5 ISA 7.12.2, 16-bit, wave32):
//  A 16x32: lane M=lane%16; halves 0..7 -> K = 8*hg+0..7, halves 8..15 ->
//           K = 16+8*hg+0..7 (hg = lane/16).
//  B 32x16: lane N=lane%16; halves 0..7 -> K = 8*hg+0..7, halves 8..15 -> K = 16+8*hg+0..7 (as A).
//  C 16x16 f32: lane N=lane%16; VGPR r -> M = r + 8*hg.
// ---------------------------------------------------------------------------
__global__ void __launch_bounds__(256)
k_gemm_min(const _Float16* __restrict__ xsT, const _Float16* __restrict__ p16,
           const float* __restrict__ x2, const float* __restrict__ p2,
           float* __restrict__ mind) {
    __shared__ _Float16 panel[2][PANELH];   // 2 x 8448 B

    const int tid   = threadIdx.x;
    const int lane  = tid & 31;
    const int wave  = tid >> 5;              // 0..7
    const int hg    = lane >> 4;
    const int lm    = lane & 15;
    const int m0    = blockIdx.x * 256 + wave * 32;
    const int chunk = blockIdx.y;
    const int b     = blockIdx.z;

    const int ptBeg = chunk ? 25 : 0;
    const int ptEnd = chunk ? NPTILES : 25;

    const _Float16* xb = xsT + (size_t)b * WHPIX * DCH;

    // Per-thread async-copy slice: 2 x 16B chunks of the 8KB panel.
    const int row0 = tid >> 5;               // chunk tid     -> row 0..7
    const int col0 = (tid & 31) << 4;        // byte col 0..496
    const int row1 = (tid + 256) >> 5;       // chunk tid+256 -> row 8..15
    const int col1 = col0;

    const uint64_t gbase = (uint64_t)(uintptr_t)xb;
    const uint32_t lbase0 = (uint32_t)(uintptr_t)&panel[0][0];
    const uint32_t lbase1 = (uint32_t)(uintptr_t)&panel[1][0];

    // --- issue async panel load for tile ptBeg into buffer 0 ---
    {
        uint64_t g0 = gbase + (uint64_t)(ptBeg * 16 + row0) * (DCH * 2) + col0;
        uint64_t g1 = gbase + (uint64_t)(ptBeg * 16 + row1) * (DCH * 2) + col1;
        uint32_t l0 = lbase0 + row0 * (LROW * 2) + col0;
        uint32_t l1 = lbase0 + row1 * (LROW * 2) + col1;
        asm volatile("global_load_async_to_lds_b128 %0, %1, off" :: "v"(l0), "v"(g0) : "memory");
        asm volatile("global_load_async_to_lds_b128 %0, %1, off" :: "v"(l1), "v"(g1) : "memory");
    }

    // Preload A fragments for both 16-proto sub-tiles: 2 x 64 VGPRs.
    v16h a0[8], a1[8];
    {
        const _Float16* ap0 = p16 + (size_t)(m0 + lm) * DCH;
        const _Float16* ap1 = p16 + (size_t)(m0 + 16 + lm) * DCH;
        #pragma unroll
        for (int s = 0; s < 8; ++s) {
            AFrag af;
            af.h[0] = *(const v8h*)(ap0 + 32 * s + 8 * hg);
            af.h[1] = *(const v8h*)(ap0 + 32 * s + 16 + 8 * hg);
            a0[s] = af.v;
            af.h[0] = *(const v8h*)(ap1 + 32 * s + 8 * hg);
            af.h[1] = *(const v8h*)(ap1 + 32 * s + 16 + 8 * hg);
            a1[s] = af.v;
        }
    }

    float p2r0[8], p2r1[8];
    #pragma unroll
    for (int r = 0; r < 8; ++r) {
        p2r0[r] = p2[m0 + r + 8 * hg];
        p2r1[r] = p2[m0 + 16 + r + 8 * hg];
    }
    const float* x2row = x2 + b * WHPIX;

    float rm0[8], rm1[8];
    #pragma unroll
    for (int r = 0; r < 8; ++r) { rm0[r] = 3.4e38f; rm1[r] = 3.4e38f; }

    // Wait for buffer 0, make visible to all waves.
    asm volatile("s_wait_asynccnt 0x0" ::: "memory");
    __syncthreads();

    int cur = 0;
    for (int pt = ptBeg; pt < ptEnd; ++pt) {
        // Prefetch next tile's panel into the other buffer.
        if (pt + 1 < ptEnd) {
            uint32_t lb = cur ? lbase0 : lbase1;
            uint64_t g0 = gbase + (uint64_t)((pt + 1) * 16 + row0) * (DCH * 2) + col0;
            uint64_t g1 = gbase + (uint64_t)((pt + 1) * 16 + row1) * (DCH * 2) + col1;
            uint32_t l0 = lb + row0 * (LROW * 2) + col0;
            uint32_t l1 = lb + row1 * (LROW * 2) + col1;
            asm volatile("global_load_async_to_lds_b128 %0, %1, off" :: "v"(l0), "v"(g0) : "memory");
            asm volatile("global_load_async_to_lds_b128 %0, %1, off" :: "v"(l1), "v"(g1) : "memory");
        }

        // Compute on current buffer: 8 K-steps, each B fragment feeds 2 WMMAs.
        const _Float16* prow = &panel[cur][lm * LROW];
        v8f c0 = {}, c1 = {};
        #pragma unroll
        for (int s = 0; s < 8; ++s) {
            AFrag bf;
            bf.h[0] = *(const v8h*)(prow + 32 * s + 8 * hg);
            bf.h[1] = *(const v8h*)(prow + 32 * s + 16 + 8 * hg);
            c0 = __builtin_amdgcn_wmma_f32_16x16x32_f16(
                    false, a0[s], false, bf.v, (short)0, c0, false, false);
            c1 = __builtin_amdgcn_wmma_f32_16x16x32_f16(
                    false, a1[s], false, bf.v, (short)0, c1, false, false);
        }

        const float x2v = x2row[pt * 16 + lm];
        #pragma unroll
        for (int r = 0; r < 8; ++r) {
            float sq0 = x2v - 2.0f * c0[r] + p2r0[r];
            float sq1 = x2v - 2.0f * c1[r] + p2r1[r];
            rm0[r] = fminf(rm0[r], sqrtf(fabsf(sq0) + 1e-14f));
            rm1[r] = fminf(rm1[r], sqrtf(fabsf(sq1) + 1e-14f));
        }

        // Next buffer's DMA done + all waves finished reading this buffer.
        asm volatile("s_wait_asynccnt 0x0" ::: "memory");
        __syncthreads();
        cur ^= 1;
    }

    // Min across the 16 lanes of each half (N dimension), per C-row r.
    #pragma unroll
    for (int r = 0; r < 8; ++r) {
        float v0 = rm0[r], v1 = rm1[r];
        #pragma unroll
        for (int off = 1; off < 16; off <<= 1) {
            v0 = fminf(v0, __shfl_xor(v0, off, 32));
            v1 = fminf(v1, __shfl_xor(v1, off, 32));
        }
        rm0[r] = v0; rm1[r] = v1;
    }
    if (lm == 0) {
        float* dst = mind + ((size_t)chunk * BATCH + b) * PPAD;
        #pragma unroll
        for (int r = 0; r < 8; ++r) {
            dst[m0 + r + 8 * hg]      = rm0[r];
            dst[m0 + 16 + r + 8 * hg] = rm1[r];
        }
    }
}

// ---------------------------------------------------------------------------
// Kernel 4: ds = softmax(leaf_params, axis=-1), 512 rows x 200 cols.
// ---------------------------------------------------------------------------
__global__ void k_softmax(const float* __restrict__ lp, float* __restrict__ ds) {
    __shared__ float red[256];
    int l = blockIdx.x, c = threadIdx.x;
    float v = (c < NCLS) ? lp[l * NCLS + c] : -3.4e38f;
    red[c] = v;
    __syncthreads();
    for (int s = 128; s > 0; s >>= 1) {
        if (c < s) red[c] = fmaxf(red[c], red[c + s]);
        __syncthreads();
    }
    float mx = red[0];
    __syncthreads();
    float e = (c < NCLS) ? expf(v - mx) : 0.0f;
    red[c] = e;
    __syncthreads();
    for (int s = 128; s > 0; s >>= 1) {
        if (c < s) red[c] += red[c + s];
        __syncthreads();
    }
    float sum = red[0];
    if (c < NCLS) ds[l * NCLS + c] = e / sum;
}

// ---------------------------------------------------------------------------
// Kernel 5: per-batch combine: min over the 2 pixel chunks -> similarities ->
// tree path products -> pa @ ds.
// ---------------------------------------------------------------------------
__global__ void k_combine(const float* __restrict__ mind, const float* __restrict__ ds,
                          float* __restrict__ out) {
    __shared__ float s_lds[NPROTO];
    __shared__ float pa_lds[NLEAF];
    int b = blockIdx.x, tid = threadIdx.x;

    for (int p = tid; p < NPROTO; p += 256) {
        float d0 = mind[(size_t)b * PPAD + p];
        float d1 = mind[((size_t)BATCH + b) * PPAD + p];
        s_lds[p] = expf(-fminf(d0, d1));
    }
    __syncthreads();

    for (int l = tid; l < NLEAF; l += 256) {
        int node = 0;
        float prod = 1.0f;
        #pragma unroll
        for (int t = 0; t < DEPTH; ++t) {
            int bit = (l >> (DEPTH - 1 - t)) & 1;
            float sv = s_lds[node];
            prod *= bit ? sv : (1.0f - sv);
            node = 2 * node + 1 + bit;
        }
        pa_lds[l] = prod;
    }
    __syncthreads();

    for (int c = tid; c < NCLS; c += 256) {
        float acc = 0.0f;
        for (int l = 0; l < NLEAF; ++l)
            acc += pa_lds[l] * ds[l * NCLS + c];
        out[b * NCLS + c] = acc;
    }
}

// ---------------------------------------------------------------------------
extern "C" void kernel_launch(void* const* d_in, const int* in_sizes, int n_in,
                              void* d_out, int out_size, void* d_ws, size_t ws_size,
                              hipStream_t stream) {
    const float* xs   = (const float*)d_in[0];   // (128,256,28,28)
    const float* prot = (const float*)d_in[1];   // (511,256,1,1)
    const float* lp   = (const float*)d_in[2];   // (512,200)
    float* out = (float*)d_out;                  // (128,200)

    // Workspace carve-up. Total ~53 MB.
    char* ws = (char*)d_ws;
    size_t off = 0;
    _Float16* xsT = (_Float16*)(ws + off); off += (size_t)BATCH * WHPIX * DCH * 2;   // 51.4 MB
    _Float16* p16 = (_Float16*)(ws + off); off += (size_t)PPAD * DCH * 2;            // 256 KB
    float* x2  = (float*)(ws + off); off += (size_t)BATCH * WHPIX * 4;               // 401 KB
    float* p2  = (float*)(ws + off); off += (size_t)PPAD * 4;
    float* mind= (float*)(ws + off); off += (size_t)NCHUNK * BATCH * PPAD * 4;       // 512 KB
    float* ds  = (float*)(ws + off); off += (size_t)NLEAF * NCLS * 4;
    (void)off; (void)ws_size; (void)in_sizes; (void)n_in; (void)out_size;

    k_protos   <<<dim3(PPAD), dim3(256), 0, stream>>>(prot, p16, p2);
    k_x2       <<<dim3((WHPIX + 255) / 256, BATCH), dim3(256), 0, stream>>>(xs, x2);
    k_transpose<<<dim3((WHPIX + 31) / 32, DCH / 32, BATCH), dim3(32, 8), 0, stream>>>(xs, xsT);
    k_softmax  <<<dim3(NLEAF), dim3(256), 0, stream>>>(lp, ds);
    k_gemm_min <<<dim3(PPAD / 256, NCHUNK, BATCH), dim3(256), 0, stream>>>(xsT, p16, x2, p2, mind);
    k_combine  <<<dim3(BATCH), dim3(256), 0, stream>>>(mind, ds, out);
}
